// MindLoss_73787538145768
// MI455X (gfx1250) — matrix operand order repaired and checked by reading, not compiled, not run on hardware
//
#include <hip/hip_runtime.h>
#include <hip/hip_bf16.h>
#include <stdint.h>

#ifndef __has_builtin
#define __has_builtin(x) 0
#endif

typedef float        v2f  __attribute__((ext_vector_type(2)));
typedef float        v8f  __attribute__((ext_vector_type(8)));
typedef _Float16     v16h __attribute__((ext_vector_type(16)));
typedef unsigned int v4u  __attribute__((ext_vector_type(4)));
typedef int          v4i  __attribute__((ext_vector_type(4)));
typedef int          v8i  __attribute__((ext_vector_type(8)));

#if __has_builtin(__builtin_amdgcn_wmma_f32_16x16x4_f32)
#define HAVE_WMMA_F32X4 1
#else
#define HAVE_WMMA_F32X4 0
#endif
#if __has_builtin(__builtin_amdgcn_tensor_load_to_lds) && __has_builtin(__builtin_amdgcn_s_wait_tensorcnt)
#define HAVE_TDM 1
#else
#define HAVE_TDM 0
#endif

#define VOL 128
#define TZ 4
#define TY 4
#define TX 16
#define IMG_MAX (10*10*22)   // (TZ+6)(TY+6)(TX+6)
#define NTILE 16384          // 32(z) * 32(y) * 8(x) * 2(batch)

// six-neighbourhood directions (dz,dy,dx) in torch index order, and the 12 pairs.
// constexpr so the fully-unrolled channel loop folds them into immediates.
constexpr int k_PI[12] = {1,2,2,3,3,4,4,4,5,5,5,5};
constexpr int k_PJ[12] = {0,0,1,0,2,1,2,3,0,1,3,4};
constexpr int k_DZ[6]  = {-1,0, 0,0,1,0};
constexpr int k_DY[6]  = { 0,0,-1,0,0,1};
constexpr int k_DX[6]  = { 0,-1,0,1,0,0};

static __device__ __forceinline__ int imax(int a,int b){return a>b?a:b;}
static __device__ __forceinline__ int imin(int a,int b){return a<b?a:b;}
static __device__ __forceinline__ int iclamp(int v,int lo,int hi){return v<lo?lo:(v>hi?hi:v);}

// Band matrix fragments for the x-pass, invariant over channels/inputs: built once per kernel.
// Band[k][n] = 1 iff k in {n, n+1, n+2}  (rowbuf col j holds x = j-1).
struct BandFrag {
#if HAVE_WMMA_F32X4
  v2f b[5];
#else
  v16h b;
#endif
};

static __device__ __forceinline__ BandFrag make_band(int hlf, int l16)
{
  BandFrag f;
#if HAVE_WMMA_F32X4
#pragma unroll
  for (int kb = 0; kb < 5; ++kb) {
    int ka = kb*4 + hlf*2;                    // f32 B 4x16: lane<16 -> K0,K1 ; lane>=16 -> K2,K3
    f.b[kb][0] = (ka     >= l16 && ka     <= l16+2) ? 1.f : 0.f;
    f.b[kb][1] = (ka + 1 >= l16 && ka + 1 <= l16+2) ? 1.f : 0.f;
  }
#else
#pragma unroll
  for (int j = 0; j < 16; ++j) {
    int k = 16*hlf + j;
    f.b[j] = (k >= l16 && k <= l16+2) ? (_Float16)1.0f : (_Float16)0.0f;
  }
#endif
  return f;
}

// ---------------- TDM tile load: 3D tile [ez][ey][ex] from a 128^3 f32 tensor ----------------
static __device__ __forceinline__ void load_tile(const float* __restrict__ src, float* img,
                                                 int zs,int ys,int xs,int ez,int ey,int ex,int lane)
{
#if HAVE_TDM
  unsigned long long ga = (unsigned long long)(uintptr_t)src
      + 4ull * ((((unsigned long long)zs)*VOL + (unsigned long long)ys)*VOL + (unsigned long long)xs);
  unsigned lds = (unsigned)(uintptr_t)img;               // low 32 bits of generic shared addr = LDS offset
  v4u g0;
  g0[0] = 1u;                                            // count=1, user descriptor
  g0[1] = lds;                                           // lds_addr
  g0[2] = (unsigned)(ga & 0xffffffffull);                // global_addr[31:0]
  g0[3] = (unsigned)((ga >> 32) & 0x1ffffffull) | (2u << 30);  // global_addr[56:32], type=2
  v8i g1;
  g1[0] = (int)(2u << 16);                               // data_size = 2 (4 bytes)
  g1[1] = (int)(128u << 16);                             // tensor_dim0[15:0] (W=128)
  g1[2] = (int)(128u << 16);                             // tensor_dim1[15:0] (H=128)
  g1[3] = (int)(((unsigned)ex) << 16);                   // tile_dim0 = ex
  g1[4] = (int)(((unsigned)ey) | (((unsigned)ez) << 16));// tile_dim1=ey, tile_dim2=ez
  g1[5] = VOL;                                           // tensor_dim0_stride = 128
  g1[6] = (int)(16384u << 16);                           // tensor_dim1_stride[15:0] = 128*128
  g1[7] = 0;
  v4i g2; g2[0] = VOL; g2[1] = 0; g2[2] = 0; g2[3] = 0;  // tensor_dim2 = 128 (D)
  v4i g3; g3[0] = 0; g3[1] = 0; g3[2] = 0; g3[3] = 0;
#if __clang_major__ >= 23
  v8i gpad; gpad[0]=0;gpad[1]=0;gpad[2]=0;gpad[3]=0;gpad[4]=0;gpad[5]=0;gpad[6]=0;gpad[7]=0;
  __builtin_amdgcn_tensor_load_to_lds(g0, g1, g2, g3, gpad, 0);
#else
  __builtin_amdgcn_tensor_load_to_lds(g0, g1, g2, g3, 0);
#endif
  __builtin_amdgcn_s_wait_tensorcnt(0);
  (void)lane;
#else
  const int tot = ez*ey*ex;
  for (int i = lane; i < tot; i += 32) {
    int lx = i % ex; int t = i / ex; int ly = t % ey; int lz = t / ey;
    img[i] = src[(((size_t)(zs+lz))*VOL + (size_t)(ys+ly))*VOL + (size_t)(xs+lx)];
  }
#endif
  __syncthreads();
}

// --------------- one MIND channel: diff2 -> separable 3^3 box; x-pass via WMMA ---------------
// Deltas arrive as compile-time constants (unrolled caller), so clamps on zero axes fold away.
static __device__ __forceinline__ void ssd_channel(
    const float* img, float* dbuf, float* zbuf, float* rowbuf,
    const BandFrag& bf,
    int lane, int hlf, int l16,
    int z0, int y0, int x0,
    int zs, int ys, int xs, int ey, int ex,
    int d1z,int d1y,int d1x, int d2z,int d2y,int d2x,
    float out[8])
{
  __syncthreads();
  // Stage A: diff2 on region [6][6][18] (tile coords offset -1), edge-clamped twice:
  // once for the diff2 (box) pad coordinate, once for the +/-2 dilated shifts.
  for (int i = lane; i < 6*6*18; i += 32) {
    int ux = i % 18; int t = i / 18; int uy = t % 6; int uz = t / 6;
    int gz = iclamp(z0 + uz - 1, 0, VOL-1);
    int gy = iclamp(y0 + uy - 1, 0, VOL-1);
    int gx = iclamp(x0 + ux - 1, 0, VOL-1);
    int az = d1z ? iclamp(gz + 2*d1z, 0, VOL-1) : gz;
    int ay = d1y ? iclamp(gy + 2*d1y, 0, VOL-1) : gy;
    int ax = d1x ? iclamp(gx + 2*d1x, 0, VOL-1) : gx;
    int bz = d2z ? iclamp(gz + 2*d2z, 0, VOL-1) : gz;
    int by = d2y ? iclamp(gy + 2*d2y, 0, VOL-1) : gy;
    int bx = d2x ? iclamp(gx + 2*d2x, 0, VOL-1) : gx;
    float a = img[((az - zs)*ey + (ay - ys))*ex + (ax - xs)];
    float b = img[((bz - zs)*ey + (by - ys))*ex + (bx - xs)];
    float d = a - b;
    dbuf[i] = d * d;
  }
  __syncthreads();
  // Stage B: z 3-tap -> zbuf [4][6][18]
  for (int i = lane; i < 4*6*18; i += 32) {
    int col = i % 108; int vz = i / 108;
    zbuf[i] = dbuf[vz*108 + col] + dbuf[(vz+1)*108 + col] + dbuf[(vz+2)*108 + col];
  }
  __syncthreads();
  // Stage C: y 3-tap -> rowbuf [16 rows][32] (cols 0..17 valid, 18..31 stay zero)
  for (int i = lane; i < 16*18; i += 32) {
    int col = i % 18; int row = i / 18;
    int vz = row >> 2, vy = row & 3;
    int b0 = (vz*6 + vy)*18 + col;
    rowbuf[row*32 + col] = zbuf[b0] + zbuf[b0 + 18] + zbuf[b0 + 36];
  }
  __syncthreads();
  // Stage D: x 3-tap as banded matmul.  out[row][n] = sum_k rowbuf[row][k] * Band[k][n].
  v8f acc = {0.f,0.f,0.f,0.f,0.f,0.f,0.f,0.f};
#if HAVE_WMMA_F32X4
#pragma unroll
  for (int kb = 0; kb < 5; ++kb) {            // K = 0..19 covers all 18 valid columns
    int ka = kb*4 + hlf*2;                    // f32 A 16x4: lane<16 -> K0,K1 ; lane>=16 -> K2,K3
    v2f A; A[0] = rowbuf[l16*32 + ka]; A[1] = rowbuf[l16*32 + ka + 1];
    acc = __builtin_amdgcn_wmma_f32_16x16x4_f32(false, A, false, bf.b[kb], (short)0, acc, false, false);
  }
#else
  v16h A;
#pragma unroll
  for (int j = 0; j < 8; ++j) {               // f16 A 16x32 layout (contiguous K: lane half h holds K 16h..16h+15)
    A[j]     = (_Float16)rowbuf[l16*32 + hlf*16 + j];
    A[8 + j] = (_Float16)rowbuf[l16*32 + hlf*16 + 8 + j];
  }
  acc = __builtin_amdgcn_wmma_f32_16x16x32_f16(false, A, false, bf.b, (short)0, acc, false, false);
#endif
  const float s = 1.0f / 27.0f;
#pragma unroll
  for (int r = 0; r < 8; ++r) out[r] = acc[r] * s;   // lane owns rows M = r + 8*hlf, x = l16
}

// --------------- full 12-channel tile: min / sum (and optional ssd store) ---------------
template<bool STORE>
static __device__ __forceinline__ void tile_compute(
    const float* __restrict__ src,
    float* img, float* dbuf, float* zbuf, float* rowbuf, float* ssdstore,
    const BandFrag& bf,
    int lane, int hlf, int l16, int z0, int y0, int x0,
    float vmin[8], float vsum[8])
{
  const int zs = imax(z0-3,0), ys = imax(y0-3,0), xs = imax(x0-3,0);
  const int ez = imin(z0+TZ+2, VOL-1) - zs + 1;
  const int ey = imin(y0+TY+2, VOL-1) - ys + 1;
  const int ex = imin(x0+TX+2, VOL-1) - xs + 1;

  load_tile(src, img, zs, ys, xs, ez, ey, ex, lane);

#pragma unroll
  for (int r = 0; r < 8; ++r) { vmin[r] = 3.402823e38f; vsum[r] = 0.f; }

#pragma unroll
  for (int c = 0; c < 12; ++c) {
    float s8[8];
    ssd_channel(img, dbuf, zbuf, rowbuf, bf, lane, hlf, l16,
                z0, y0, x0, zs, ys, xs, ey, ex,
                k_DZ[k_PI[c]], k_DY[k_PI[c]], k_DX[k_PI[c]],
                k_DZ[k_PJ[c]], k_DY[k_PJ[c]], k_DX[k_PJ[c]], s8);
#pragma unroll
    for (int r = 0; r < 8; ++r) {
      vmin[r] = fminf(vmin[r], s8[r]);
      vsum[r] += s8[r];
      if (STORE) ssdstore[c*256 + (r + 8*hlf)*16 + l16] = s8[r];
    }
  }
}

// --------------- pass 1: per-input sum of mind_var (for the global mean m) ---------------
__global__ void __launch_bounds__(32)
mind_pass1(const float* __restrict__ P, const float* __restrict__ T, float* __restrict__ ws)
{
  __shared__ __align__(16) float img[IMG_MAX];
  __shared__ float dbuf[6*6*18];
  __shared__ float zbuf[4*6*18];
  __shared__ float rowbuf[16*32];

  const int lane = threadIdx.x, hlf = lane >> 4, l16 = lane & 15;
  const unsigned tid = blockIdx.x;
  const int xt = tid & 7, yt = (tid>>3)&31, zt = (tid>>8)&31, n = (tid>>13)&1;
  const float* src = (blockIdx.y ? T : P) + (size_t)n * (VOL*VOL*VOL);
  const int z0 = zt*TZ, y0 = yt*TY, x0 = xt*TX;

  const BandFrag bf = make_band(hlf, l16);
  for (int i = lane; i < 16*32; i += 32) rowbuf[i] = 0.f;

  float vmin[8], vsum[8];
  tile_compute<false>(src, img, dbuf, zbuf, rowbuf, nullptr, bf,
                      lane, hlf, l16, z0, y0, x0, vmin, vsum);

  float part = 0.f;
#pragma unroll
  for (int r = 0; r < 8; ++r) part += vsum[r]*(1.f/12.f) - vmin[r];   // mean(ssd)-min = mind_var
  for (int off = 16; off > 0; off >>= 1) part += __shfl_xor(part, off);
  if (lane == 0) atomicAdd(&ws[blockIdx.y], part);
}

// --------------- pass 2: recompute both descriptors, clamp with m, accumulate MSE ---------------
__global__ void __launch_bounds__(32)
mind_pass2(const float* __restrict__ P, const float* __restrict__ T, float* __restrict__ ws)
{
  __shared__ __align__(16) float img[IMG_MAX];
  __shared__ float dbuf[6*6*18];
  __shared__ float zbuf[4*6*18];
  __shared__ float rowbuf[16*32];
  __shared__ float ssdb[2*12*256];

  const int lane = threadIdx.x, hlf = lane >> 4, l16 = lane & 15;
  const unsigned tid = blockIdx.x;
  const int xt = tid & 7, yt = (tid>>3)&31, zt = (tid>>8)&31, n = (tid>>13)&1;
  const int z0 = zt*TZ, y0 = yt*TY, x0 = xt*TX;
  const size_t nb = (size_t)n * (VOL*VOL*VOL);

  const BandFrag bf = make_band(hlf, l16);
  for (int i = lane; i < 16*32; i += 32) rowbuf[i] = 0.f;

  float vminA[2][8], vsumA[2][8];
#pragma unroll
  for (int inp = 0; inp < 2; ++inp) {
    const float* src = (inp ? T : P) + nb;
    tile_compute<true>(src, img, dbuf, zbuf, rowbuf, ssdb + inp*12*256, bf,
                       lane, hlf, l16, z0, y0, x0, vminA[inp], vsumA[inp]);
  }

  const float m0 = ws[0] * (1.f/4194304.f);   // mean over 2*128^3
  const float m1 = ws[1] * (1.f/4194304.f);

  float part = 0.f;
#pragma unroll
  for (int r = 0; r < 8; ++r) {
    const int vox = (r + 8*hlf)*16 + l16;
    float vP = fminf(fmaxf(vsumA[0][r]*(1.f/12.f) - vminA[0][r], m0*0.001f), m0*1000.f);
    float vT = fminf(fmaxf(vsumA[1][r]*(1.f/12.f) - vminA[1][r], m1*0.001f), m1*1000.f);
    float rP = -1.f / vP, rT = -1.f / vT;
#pragma unroll
    for (int c = 0; c < 12; ++c) {
      float ep = __expf((ssdb[c*256 + vox]        - vminA[0][r]) * rP);
      float et = __expf((ssdb[(12 + c)*256 + vox] - vminA[1][r]) * rT);
      float d = ep - et;
      part += d * d;
    }
  }
  for (int off = 16; off > 0; off >>= 1) part += __shfl_xor(part, off);
  if (lane == 0) atomicAdd(&ws[2], part);
}

__global__ void mind_init(float* __restrict__ ws) { ws[0] = 0.f; ws[1] = 0.f; ws[2] = 0.f; }
__global__ void mind_final(const float* __restrict__ ws, float* __restrict__ out)
{
  out[0] = ws[2] * (1.0f / 50331648.0f);      // mean over 2*12*128^3, LOSS_WEIGHT = 1
}

extern "C" void kernel_launch(void* const* d_in, const int* in_sizes, int n_in,
                              void* d_out, int out_size, void* d_ws, size_t ws_size,
                              hipStream_t stream) {
  (void)in_sizes; (void)n_in; (void)out_size; (void)ws_size;
  const float* P = (const float*)d_in[0];
  const float* T = (const float*)d_in[1];
  float* ws  = (float*)d_ws;
  float* out = (float*)d_out;

  mind_init<<<dim3(1), dim3(1), 0, stream>>>(ws);
  mind_pass1<<<dim3(NTILE, 2), dim3(32), 0, stream>>>(P, T, ws);
  mind_pass2<<<dim3(NTILE), dim3(32), 0, stream>>>(P, T, ws);
  mind_final<<<dim3(1), dim3(1), 0, stream>>>(ws, out);
}
